// Depth_MoE_84318797955382
// MI455X (gfx1250) — matrix operand order repaired and checked by reading, not compiled, not run on hardware
//
#include <hip/hip_runtime.h>
#include <math.h>

typedef __attribute__((ext_vector_type(16))) _Float16 v16h;
typedef __attribute__((ext_vector_type(8)))  float    v8f;

#define BB 8
#define CC 19
#define HH 480
#define WW 480
#define HWPIX (HH*WW)
#define HQ 160
#define WQ 160
#define NPATCH (BB*HQ*WQ)
#define OUTMAP (BB*HWPIX)
#define BPS 128            // blocks per sample in pre-pass

// Wave-local LDS ordering: all per-patch scratch is wave-private and wave32
// lanes run in lockstep; LDS ops from one wave execute in order (DScnt).
// A dscnt drain + compiler memory barrier is sufficient -- no block barrier.
#define WSYNC() asm volatile("s_wait_dscnt 0" ::: "memory")

// ---------------- fragment helpers (ISA 7.12.2 layouts, wave32) -------------

__device__ __forceinline__ v16h frag_load_a(const _Float16* st, int lane) {
  // A 16x32 f16: lanes 0-15 row m=lane, K 0-7 (v0..3) & 16-23 (v4..7);
  // lanes 16-31 row m=lane-16, K 8-15 & 24-31.  2 f16 per VGPR.
  int m = lane & 15, kh = lane >> 4;
  v16h a;
#pragma unroll
  for (int v = 0; v < 8; ++v) {
    int kb = ((v < 4) ? (2 * v) : (16 + 2 * (v - 4))) + 8 * kh;
    a[2 * v]     = st[m * 32 + kb];
    a[2 * v + 1] = st[m * 32 + kb + 1];
  }
  return a;
}

__device__ __forceinline__ v16h frag_load_b(const _Float16* st, int lane) {
  // B 32x16 f16 staged TRANSPOSED in LDS as [N][K] (K-major rows).
  // lanes 0-15: col n=lane, K 0-7 & 16-23 ; lanes 16-31: col n=lane-16, K 8-15 & 24-31.
  int n = lane & 15, kh = lane >> 4;
  v16h b;
#pragma unroll
  for (int v = 0; v < 8; ++v) {
    int kb = ((v < 4) ? (2 * v) : (16 + 2 * (v - 4))) + 8 * kh;
    b[2 * v]     = st[n * 32 + kb];
    b[2 * v + 1] = st[n * 32 + kb + 1];
  }
  return b;
}

__device__ __forceinline__ v8f wmma16(v16h a, v16h b, v8f c) {
  return __builtin_amdgcn_wmma_f32_16x16x32_f16(false, a, false, b, (short)0, c,
                                                false, false);
}

__device__ __forceinline__ float gelu_f(float x) {
  return 0.5f * x * (1.f + erff(x * 0.70710678118654752f));
}

// ---------------- pass 1: sem softmax-max + per-sample moments --------------

__global__ __launch_bounds__(256) void pre_kernel(const float* __restrict__ depth,
                                                  const float* __restrict__ sem,
                                                  float* __restrict__ semout,
                                                  float* __restrict__ accum) {
  int b = blockIdx.x / BPS;
  int blk = blockIdx.x % BPS;
  int tid = threadIdx.x;
  float sd = 0.f, sdq = 0.f, ss = 0.f, ssq = 0.f;
  for (int p = blk * 256 + tid; p < HWPIX; p += BPS * 256) {
    int gi = b * HWPIX + p;
    // keep the HBM stream ahead of the 19-channel scan (global_prefetch_b8)
    int pn = p + BPS * 256;
    if (pn < HWPIX) {
      __builtin_prefetch(&depth[b * HWPIX + pn], 0, 1);
      __builtin_prefetch(&sem[(b * CC) * HWPIX + pn], 0, 1);
      __builtin_prefetch(&sem[(b * CC + 9) * HWPIX + pn], 0, 1);
    }
    float d = depth[gi];
    sd += d; sdq += d * d;
    float vv[CC];
    float mx = -1e30f;
#pragma unroll
    for (int c = 0; c < CC; ++c) {
      vv[c] = sem[(b * CC + c) * HWPIX + p];
      mx = fmaxf(mx, vv[c]);
    }
    float s = 0.f;
#pragma unroll
    for (int c = 0; c < CC; ++c) s += __expf(vv[c] - mx);
    float prob = 1.f / s;            // max of softmax
    semout[gi] = prob;
    ss += prob; ssq += prob * prob;
  }
  __shared__ float red[4][256];
  red[0][tid] = sd; red[1][tid] = sdq; red[2][tid] = ss; red[3][tid] = ssq;
  __syncthreads();
  for (int off = 128; off > 0; off >>= 1) {
    if (tid < off)
      for (int q = 0; q < 4; ++q) red[q][tid] += red[q][tid + off];
    __syncthreads();
  }
  if (tid == 0)
    for (int q = 0; q < 4; ++q) atomicAdd(&accum[b * 4 + q], red[q][0]);
}

__global__ void init_kernel(float* ws) {
  int t = threadIdx.x;
  if (t < 80) ws[t] = 0.f;
}

__global__ void fin_kernel(const float* __restrict__ accum, float* __restrict__ st) {
  int b = threadIdx.x;
  if (b < BB) {
    float n = (float)HWPIX;
    float s = accum[b * 4 + 0], sq = accum[b * 4 + 1];
    float m = s / n;
    float var = fmaxf((sq - s * s / n) / (n - 1.f), 0.f);
    st[b * 4 + 0] = m;
    st[b * 4 + 1] = 1.f / (sqrtf(var) + 1e-6f);
    s = accum[b * 4 + 2]; sq = accum[b * 4 + 3];
    m = s / n;
    var = fmaxf((sq - s * s / n) / (n - 1.f), 0.f);
    st[b * 4 + 2] = m;
    st[b * 4 + 3] = 1.f / (sqrtf(var) + 1e-6f);
  }
}

__global__ void avg_kernel(const float* __restrict__ gate, float* __restrict__ out) {
  int t = threadIdx.x;
  if (t < 3) out[OUTMAP + t] = gate[t] * (1.f / (float)NPATCH);
}

// ---------------- main per-patch kernel (1 wave = 1 patch) ------------------

struct Params {
  float dp_w[16], dp_b[16], sp_w[16], sp_b[16];
  float in_proj_b[48], out_b[16];
  float n1_g[16], n1_b[16];
  float eg_b1[16], eg_b2[16], es_b1[16], es_b2[16], ef_b1[16], ef_b2[16];
  float n2_g[16], n2_b[16];
  float g_w1[48 * 16], g_b1[16], g_w2[48], g_b2[3];
  float po_w[16], po_b0;
};

struct WaveScratch {
  _Float16 stage[16 * 32];   // A/B staging (row- or N-major, K stride 32)
  float t0[16 * 48];         // qkv, later expert outputs (3 slabs)
  float t1[16 * 16];         // scores / hidden / generic D target
  float x[16 * 16];          // LN'd token features
  float df[16 * 16];         // depth_feat
  float sf[16 * 16];         // sem_feat
  float ao[16 * 16];         // attention output
  float misc[64];            // pixels, stats, gate
};

__device__ __forceinline__ v16h build_wfrag(_Float16* stage, int lane,
                                            const float* W, int ld, int colOff) {
  WSYNC();
  for (int i = lane; i < 512; i += 32) {
    int n = i >> 5, k = i & 31;
    float v = (k < 16) ? W[k * ld + colOff + n] : 0.f;
    stage[i] = (_Float16)v;
  }
  WSYNC();
  return frag_load_b(stage, lane);
}

__global__ __launch_bounds__(128) void moe_patch_kernel(
    const float* __restrict__ depth, const float* __restrict__ semn,
    const float* __restrict__ stats4,
    const float* __restrict__ dp_w, const float* __restrict__ dp_b,
    const float* __restrict__ sp_w, const float* __restrict__ sp_b,
    const float* __restrict__ in_proj_w, const float* __restrict__ in_proj_b,
    const float* __restrict__ out_w, const float* __restrict__ out_b,
    const float* __restrict__ n1_g, const float* __restrict__ n1_b,
    const float* __restrict__ eg_w1, const float* __restrict__ eg_b1,
    const float* __restrict__ eg_w2, const float* __restrict__ eg_b2,
    const float* __restrict__ es_w1, const float* __restrict__ es_b1,
    const float* __restrict__ es_w2, const float* __restrict__ es_b2,
    const float* __restrict__ ef_w1, const float* __restrict__ ef_b1,
    const float* __restrict__ ef_w2, const float* __restrict__ ef_b2,
    const float* __restrict__ n2_g, const float* __restrict__ n2_b,
    const float* __restrict__ g_w1, const float* __restrict__ g_b1,
    const float* __restrict__ g_w2, const float* __restrict__ g_b2,
    const float* __restrict__ po_w, const float* __restrict__ po_b,
    float* __restrict__ outMap, float* __restrict__ gate_accum) {
  __shared__ Params P;
  __shared__ WaveScratch WS[4];

  {
    int tid = threadIdx.x;
    for (int i = tid; i < 16; i += 128) {
      P.dp_w[i] = dp_w[i];  P.dp_b[i] = dp_b[i];
      P.sp_w[i] = sp_w[i];  P.sp_b[i] = sp_b[i];
      P.out_b[i] = out_b[i];
      P.n1_g[i] = n1_g[i];  P.n1_b[i] = n1_b[i];
      P.eg_b1[i] = eg_b1[i]; P.eg_b2[i] = eg_b2[i];
      P.es_b1[i] = es_b1[i]; P.es_b2[i] = es_b2[i];
      P.ef_b1[i] = ef_b1[i]; P.ef_b2[i] = ef_b2[i];
      P.n2_g[i] = n2_g[i];  P.n2_b[i] = n2_b[i];
      P.g_b1[i] = g_b1[i];  P.po_w[i] = po_w[i];
    }
    for (int i = tid; i < 48; i += 128) { P.in_proj_b[i] = in_proj_b[i]; P.g_w2[i] = g_w2[i]; }
    for (int i = tid; i < 768; i += 128) P.g_w1[i] = g_w1[i];
    if (tid < 3) P.g_b2[tid] = g_b2[tid];
    if (tid == 0) P.po_b0 = po_b[0];
  }
  __syncthreads();   // the only block-wide barrier: params are shared

  const int lane = threadIdx.x & 31;
  const int wv = threadIdx.x >> 5;
  WaveScratch& S = WS[wv];

  const int pid = blockIdx.x * 4 + wv;
  const int b = pid / (HQ * WQ);
  const int r0 = pid % (HQ * WQ);
  const int ph = r0 / WQ, pw = r0 % WQ;

  // ---- weight B fragments (held in VGPRs) ----
  v16h fip0 = build_wfrag(S.stage, lane, in_proj_w, 48, 0);
  v16h fip1 = build_wfrag(S.stage, lane, in_proj_w, 48, 16);
  v16h fip2 = build_wfrag(S.stage, lane, in_proj_w, 48, 32);
  v16h fow  = build_wfrag(S.stage, lane, out_w, 16, 0);
  v16h feg1 = build_wfrag(S.stage, lane, eg_w1, 16, 0);
  v16h feg2 = build_wfrag(S.stage, lane, eg_w2, 16, 0);
  v16h fes1 = build_wfrag(S.stage, lane, es_w1, 16, 0);
  v16h fes2 = build_wfrag(S.stage, lane, es_w2, 16, 0);
  v16h fef1 = build_wfrag(S.stage, lane, ef_w1, 16, 0);
  v16h fef2 = build_wfrag(S.stage, lane, ef_w2, 16, 0);

  // ---- generic helpers (wave-local sync only) ----
  auto buildA = [&](auto f) -> v16h {   // f(t,k) -> float, A is [16 x 32]
    WSYNC();
    for (int i = lane; i < 512; i += 32) {
      int t = i >> 5, k = i & 31;
      S.stage[i] = (_Float16)f(t, k);
    }
    WSYNC();
    return frag_load_a(S.stage, lane);
  };
  auto buildB = [&](auto f) -> v16h {   // f(n,k) -> float, staged [N][K]
    WSYNC();
    for (int i = lane; i < 512; i += 32) {
      int n = i >> 5, k = i & 31;
      S.stage[i] = (_Float16)f(n, k);
    }
    WSYNC();
    return frag_load_b(S.stage, lane);
  };
  auto storeD = [&](v8f d, auto f) {    // f(m,n,val) writes LDS
    int n = lane & 15, mb = (lane >> 4) * 8;
#pragma unroll
    for (int r = 0; r < 8; ++r) f(mb + r, n, d[r]);
    WSYNC();
  };

  // ---- normalized input pixels ----
  const float meanD = stats4[b * 4 + 0], rcpD = stats4[b * 4 + 1];
  const float meanS = stats4[b * 4 + 2], rcpS = stats4[b * 4 + 3];
  if (lane < 9) {
    int rr = ph * 3 + lane / 3, cc = pw * 3 + lane % 3;
    int gi = b * HWPIX + rr * WW + cc;
    S.misc[lane] = (depth[gi] - meanD) * rcpD;
    S.misc[16 + lane] = (semn[gi] - meanS) * rcpS;
  }
  WSYNC();

  // ---- depth_feat / sem_feat (rank-1) ----
  for (int i = lane; i < 256; i += 32) {
    int t = i >> 4, e = i & 15;
    float dv = 0.f, sv = 0.f;
    if (t < 9) {
      dv = S.misc[t] * P.dp_w[e] + P.dp_b[e];
      sv = S.misc[16 + t] * P.sp_w[e] + P.sp_b[e];
    }
    S.df[i] = dv; S.sf[i] = sv;
  }
  WSYNC();

  // ---- x = LN1(df + sf) ----
  if (lane < 16) {
    int t = lane;
    if (t < 9) {
      float m = 0.f;
      for (int e = 0; e < 16; ++e) m += S.df[t * 16 + e] + S.sf[t * 16 + e];
      m *= (1.f / 16.f);
      float va = 0.f;
      for (int e = 0; e < 16; ++e) {
        float d = S.df[t * 16 + e] + S.sf[t * 16 + e] - m;
        va += d * d;
      }
      va *= (1.f / 16.f);
      float rs = rsqrtf(va + 1e-5f);
      for (int e = 0; e < 16; ++e) {
        float d = S.df[t * 16 + e] + S.sf[t * 16 + e] - m;
        S.x[t * 16 + e] = d * rs * P.n1_g[e] + P.n1_b[e];
      }
    } else {
      for (int e = 0; e < 16; ++e) S.x[t * 16 + e] = 0.f;
    }
  }
  WSYNC();

  // ---- qkv = x @ in_proj + b : 3 WMMAs into t0[16x48] ----
  {
    v16h ax = buildA([&](int t, int k) {
      return (t < 9 && k < 16) ? S.x[t * 16 + k] : 0.f;
    });
    v8f cz = {};
    v8f dq = wmma16(ax, fip0, cz);
    storeD(dq, [&](int m, int n, float v) { S.t0[m * 48 + n] = v + P.in_proj_b[n]; });
    v8f dk = wmma16(ax, fip1, cz);
    storeD(dk, [&](int m, int n, float v) { S.t0[m * 48 + 16 + n] = v + P.in_proj_b[16 + n]; });
    v8f dv = wmma16(ax, fip2, cz);
    storeD(dv, [&](int m, int n, float v) { S.t0[m * 48 + 32 + n] = v + P.in_proj_b[32 + n]; });
  }

  // ---- 2-head 9-token attention ----
  const float scl = 0.35355339059327f;  // 1/sqrt(8)
  for (int hd2 = 0; hd2 < 2; ++hd2) {
    v16h aq = buildA([&](int t, int k) {
      return (t < 9 && k < 8) ? S.t0[t * 48 + hd2 * 8 + k] : 0.f;
    });
    v16h bk = buildB([&](int n, int k) {
      return (n < 9 && k < 8) ? S.t0[n * 48 + 16 + hd2 * 8 + k] : 0.f;
    });
    v8f cz = {};
    v8f sc = wmma16(aq, bk, cz);
    storeD(sc, [&](int m, int n, float v) { S.t1[m * 16 + n] = v * scl; });
    if (lane < 9) {  // row softmax over 9 keys
      int t = lane;
      float mx = -1e30f;
      for (int k = 0; k < 9; ++k) mx = fmaxf(mx, S.t1[t * 16 + k]);
      float sum = 0.f;
      for (int k = 0; k < 9; ++k) {
        float e = __expf(S.t1[t * 16 + k] - mx);
        S.t1[t * 16 + k] = e; sum += e;
      }
      float rs = 1.f / sum;
      for (int k = 0; k < 9; ++k) S.t1[t * 16 + k] *= rs;
    }
    WSYNC();
    v16h aa = buildA([&](int t, int k) {
      return (t < 9 && k < 9) ? S.t1[t * 16 + k] : 0.f;
    });
    v16h bv = buildB([&](int n, int k) {
      return (n < 8 && k < 9) ? S.t0[k * 48 + 32 + hd2 * 8 + n] : 0.f;
    });
    v8f oo = wmma16(aa, bv, cz);
    storeD(oo, [&](int m, int n, float v) {
      if (n < 8) S.ao[m * 16 + hd2 * 8 + n] = v;
    });
  }

  // ---- out projection + x = LN1(x + ao@out_w + out_b) ----
  {
    v16h aao = buildA([&](int t, int k) {
      return (t < 9 && k < 16) ? S.ao[t * 16 + k] : 0.f;
    });
    v8f cz = {};
    v8f dop = wmma16(aao, fow, cz);
    storeD(dop, [&](int m, int n, float v) { S.t1[m * 16 + n] = v + P.out_b[n]; });
    if (lane < 9) {
      int t = lane;
      float m = 0.f;
      for (int e = 0; e < 16; ++e) m += S.x[t * 16 + e] + S.t1[t * 16 + e];
      m *= (1.f / 16.f);
      float va = 0.f;
      for (int e = 0; e < 16; ++e) {
        float d = S.x[t * 16 + e] + S.t1[t * 16 + e] - m;
        va += d * d;
      }
      va *= (1.f / 16.f);
      float rs = rsqrtf(va + 1e-5f);
      for (int e = 0; e < 16; ++e) {
        float d = S.x[t * 16 + e] + S.t1[t * 16 + e] - m;
        S.x[t * 16 + e] = d * rs * P.n1_g[e] + P.n1_b[e];
      }
    }
    WSYNC();
  }

  // ---- experts: o = in + gelu(in@w1+b1)@w2+b2 into t0 slabs ----
  auto expert = [&](v16h w1, const float* b1, v16h w2, const float* b2,
                    int mode, int slab) {
    auto inval = [&](int t, int k) -> float {
      float v = S.x[t * 16 + k];
      if (mode != 1) v += S.df[t * 16 + k];
      if (mode != 0) v += S.sf[t * 16 + k];
      return v;
    };
    v16h a1 = buildA([&](int t, int k) {
      return (t < 9 && k < 16) ? inval(t, k) : 0.f;
    });
    v8f cz = {};
    v8f h = wmma16(a1, w1, cz);
    storeD(h, [&](int m, int n, float v) { S.t1[m * 16 + n] = gelu_f(v + b1[n]); });
    v16h a2 = buildA([&](int t, int k) {
      return (t < 9 && k < 16) ? S.t1[t * 16 + k] : 0.f;
    });
    v8f o = wmma16(a2, w2, cz);
    storeD(o, [&](int m, int n, float v) {
      S.t0[m * 48 + slab * 16 + n] = (m < 9) ? inval(m, n) + v + b2[n] : 0.f;
    });
  };
  expert(feg1, P.eg_b1, feg2, P.eg_b2, 0, 0);
  expert(fes1, P.es_b1, fes2, P.es_b2, 1, 1);
  expert(fef1, P.ef_b1, fef2, P.ef_b2, 2, 2);

  // ---- gate: stats = [mean, max, std(ddof=1)] over 9 tokens ----
  if (lane < 16) {
    int e = lane;
    float m = 0.f, mx = -1e30f;
    for (int t = 0; t < 9; ++t) {
      float v = S.x[t * 16 + e];
      m += v; mx = fmaxf(mx, v);
    }
    m *= (1.f / 9.f);
    float va = 0.f;
    for (int t = 0; t < 9; ++t) {
      float d = S.x[t * 16 + e] - m;
      va += d * d;
    }
    va *= (1.f / 8.f);
    S.misc[e] = m; S.misc[16 + e] = mx; S.misc[32 + e] = sqrtf(va);
  }
  WSYNC();
  if (lane < 16) {
    int j = lane;
    float acc = P.g_b1[j];
    for (int i = 0; i < 48; ++i) acc += S.misc[i] * P.g_w1[i * 16 + j];
    S.misc[48 + j] = gelu_f(acc);
  }
  WSYNC();
  if (lane == 0) {
    float z[3];
    float mx = -1e30f;
    for (int l = 0; l < 3; ++l) {
      z[l] = P.g_b2[l];
      for (int j = 0; j < 16; ++j) z[l] += S.misc[48 + j] * P.g_w2[j * 3 + l];
      mx = fmaxf(mx, z[l]);
    }
    float s = 0.f;
    for (int l = 0; l < 3; ++l) { z[l] = __expf(z[l] - mx); s += z[l]; }
    float rs = 1.f / s;
    for (int l = 0; l < 3; ++l) S.misc[52 + l] = z[l] * rs;
  }
  WSYNC();
  if (lane < 3) atomicAdd(&gate_accum[lane], S.misc[52 + lane]);
  const float g0 = S.misc[52], g1 = S.misc[53], g2 = S.misc[54];

  // ---- moe combine + LN2 + sigmoid head, write output pixels ----
  if (lane < 9) {
    int t = lane;
    float m = 0.f;
    for (int e = 0; e < 16; ++e) {
      float s = S.t0[t * 48 + e] * g0 + S.t0[t * 48 + 16 + e] * g1 +
                S.t0[t * 48 + 32 + e] * g2 + S.x[t * 16 + e];
      m += s;
    }
    m *= (1.f / 16.f);
    float va = 0.f;
    for (int e = 0; e < 16; ++e) {
      float s = S.t0[t * 48 + e] * g0 + S.t0[t * 48 + 16 + e] * g1 +
                S.t0[t * 48 + 32 + e] * g2 + S.x[t * 16 + e];
      float d = s - m;
      va += d * d;
    }
    va *= (1.f / 16.f);
    float rs = rsqrtf(va + 1e-5f);
    float acc = P.po_b0;
    for (int e = 0; e < 16; ++e) {
      float s = S.t0[t * 48 + e] * g0 + S.t0[t * 48 + 16 + e] * g1 +
                S.t0[t * 48 + 32 + e] * g2 + S.x[t * 16 + e];
      float ln = (s - m) * rs * P.n2_g[e] + P.n2_b[e];
      acc += ln * P.po_w[e];
    }
    float y = 1.f / (1.f + __expf(-acc));
    int rr = ph * 3 + t / 3, cc = pw * 3 + t % 3;
    outMap[b * HWPIX + rr * WW + cc] = y;
  }
}

// ---------------- launcher --------------------------------------------------

extern "C" void kernel_launch(void* const* d_in, const int* in_sizes, int n_in,
                              void* d_out, int out_size, void* d_ws, size_t ws_size,
                              hipStream_t stream) {
  const float* depth      = (const float*)d_in[0];
  const float* sem        = (const float*)d_in[1];
  const float* dp_w       = (const float*)d_in[2];
  const float* dp_b       = (const float*)d_in[3];
  const float* sp_w       = (const float*)d_in[4];
  const float* sp_b       = (const float*)d_in[5];
  const float* in_proj_w  = (const float*)d_in[6];
  const float* in_proj_b  = (const float*)d_in[7];
  const float* out_w      = (const float*)d_in[8];
  const float* out_b      = (const float*)d_in[9];
  const float* n1_g       = (const float*)d_in[10];
  const float* n1_b       = (const float*)d_in[11];
  const float* eg_w1      = (const float*)d_in[12];
  const float* eg_b1      = (const float*)d_in[13];
  const float* eg_w2      = (const float*)d_in[14];
  const float* eg_b2      = (const float*)d_in[15];
  const float* es_w1      = (const float*)d_in[16];
  const float* es_b1      = (const float*)d_in[17];
  const float* es_w2      = (const float*)d_in[18];
  const float* es_b2      = (const float*)d_in[19];
  const float* ef_w1      = (const float*)d_in[20];
  const float* ef_b1      = (const float*)d_in[21];
  const float* ef_w2      = (const float*)d_in[22];
  const float* ef_b2      = (const float*)d_in[23];
  const float* n2_g       = (const float*)d_in[24];
  const float* n2_b       = (const float*)d_in[25];
  const float* g_w1       = (const float*)d_in[26];
  const float* g_b1       = (const float*)d_in[27];
  const float* g_w2       = (const float*)d_in[28];
  const float* g_b2       = (const float*)d_in[29];
  const float* po_w       = (const float*)d_in[30];
  const float* po_b       = (const float*)d_in[31];

  float* ws     = (float*)d_ws;
  float* accum  = ws;        // 32 floats: per-sample sum/sumsq (depth, sem)
  float* gatea  = ws + 32;   // 3 floats: gate accumulator
  float* stats4 = ws + 40;   // 32 floats: mean / 1/(std+eps)
  float* semBuf = ws + 128;  // B*H*W floats: max(softmax(sem))
  float* out    = (float*)d_out;

  init_kernel<<<1, 128, 0, stream>>>(ws);
  pre_kernel<<<dim3(BB * BPS), 256, 0, stream>>>(depth, sem, semBuf, accum);
  fin_kernel<<<1, 64, 0, stream>>>(accum, stats4);
  moe_patch_kernel<<<dim3(NPATCH / 4), 128, 0, stream>>>(
      depth, semBuf, stats4,
      dp_w, dp_b, sp_w, sp_b, in_proj_w, in_proj_b, out_w, out_b, n1_g, n1_b,
      eg_w1, eg_b1, eg_w2, eg_b2, es_w1, es_b1, es_w2, es_b2,
      ef_w1, ef_b1, ef_w2, ef_b2, n2_g, n2_b, g_w1, g_b1, g_w2, g_b2,
      po_w, po_b, out, gatea);
  avg_kernel<<<1, 32, 0, stream>>>(gatea, out);
}
